// MultiHeadSelfAttention_21998822490501
// MI455X (gfx1250) — matrix operand order repaired and checked by reading, not compiled, not run on hardware
//
#include <hip/hip_runtime.h>
#include <cstdint>

typedef _Float16 half_t;
typedef __attribute__((ext_vector_type(16))) _Float16 v16h;
typedef __attribute__((ext_vector_type(8)))  _Float16 v8h;
typedef __attribute__((ext_vector_type(4)))  _Float16 v4h;
typedef __attribute__((ext_vector_type(8)))  float    v8f;

#define B_DIM 4
#define T_DIM 2048
#define C_DIM 1024
#define H_DIM 16
#define D_DIM 64
#define N3C   3072
#define M_DIM (B_DIM * T_DIM)   // 8192

static __device__ __forceinline__ uint32_t lds_offset(const void* p) {
    // LDS aperture generic address: addr[31:0] is the LDS byte offset.
    return (uint32_t)(uintptr_t)p;
}

// ---------------------------------------------------------------------------
// Kernel 1: qkv = x @ W_t + b -> q/k/v ws in [B,H,T,D] f16.
// 256 threads = 8 waves; block tile 64(M) x 128(N); wave tile 32x32
// (2x2 wmma tiles), K-step 32, v_wmma_f32_16x16x32_f16.
// ---------------------------------------------------------------------------
__global__ __launch_bounds__(256) void qkv_gemm_kernel(
    const float* __restrict__ x, const float* __restrict__ W,
    const float* __restrict__ bias,
    half_t* __restrict__ qws, half_t* __restrict__ kws, half_t* __restrict__ vws)
{
    __shared__ __attribute__((aligned(64))) half_t Al[64 * 32];    // [m][k]
    __shared__ __attribute__((aligned(64))) half_t Bl[128 * 32];   // [n][k] (transposed)

    const int tid  = threadIdx.x;
    const int lane = tid & 31;
    const int wave = tid >> 5;
    const int m0 = blockIdx.x * 64;
    const int n0 = blockIdx.y * 128;
    const int wm = (wave >> 2) * 32;   // 0 or 32
    const int wn = (wave & 3) * 32;    // 0,32,64,96

    const int kbaseA = (lane < 16) ? 0 : 16;   // A-fragment K run base
    const int kloB   = (lane < 16) ? 0 : 16;   // B-fragment K base
    const int lane16 = lane & 15;

    // staging coordinates (compile-time trip counts, vector accesses)
    const int ar0 = tid >> 3;            // A: rows for the 2 float4s
    const int ac4 = (tid & 7) * 4;       // A: col group
    const int bn4 = (lane) * 4 & 127;    // B: 4 consecutive n  (tid&31)*4
    const int bk  = (tid >> 5) * 4;      // B: 4 consecutive k

    v8f acc[2][2] = {};

    for (int k0 = 0; k0 < C_DIM; k0 += 32) {
        __syncthreads();
        // --- stage A tile 64x32: 2 x float4 per thread, b128 loads ---
#pragma unroll
        for (int it = 0; it < 2; ++it) {
            const int r = ar0 + it * 32;
            const float4 f = *(const float4*)&x[(size_t)(m0 + r) * C_DIM + k0 + ac4];
            v4h h; h[0] = (half_t)f.x; h[1] = (half_t)f.y;
                   h[2] = (half_t)f.z; h[3] = (half_t)f.w;
            *(v4h*)&Al[r * 32 + ac4] = h;
        }
        // --- stage B tile 32x128 transposed: 4 x float4 loads, 4 x b64 stores ---
        {
            float4 w4[4];
#pragma unroll
            for (int j = 0; j < 4; ++j)
                w4[j] = *(const float4*)&W[(size_t)(k0 + bk + j) * N3C + n0 + bn4];
#pragma unroll
            for (int nn = 0; nn < 4; ++nn) {
                v4h h;
#pragma unroll
                for (int j = 0; j < 4; ++j)
                    h[j] = (half_t)((&w4[j].x)[nn]);
                *(v4h*)&Bl[(bn4 + nn) * 32 + bk] = h;
            }
        }
        __syncthreads();

        v16h a[2], bf[2];
#pragma unroll
        for (int r = 0; r < 2; ++r) {
            const half_t* ap = &Al[(wm + r * 16 + lane16) * 32 + kbaseA];
            v8h lo = *(const v8h*)ap;
            v8h hi = *(const v8h*)(ap + 8);
#pragma unroll
            for (int i = 0; i < 8; ++i) { a[r][i] = lo[i]; a[r][i + 8] = hi[i]; }
        }
#pragma unroll
        for (int c = 0; c < 2; ++c)
            bf[c] = *(const v16h*)&Bl[(wn + c * 16 + lane16) * 32 + kloB];

#pragma unroll
        for (int r = 0; r < 2; ++r)
#pragma unroll
            for (int c = 0; c < 2; ++c)
                acc[r][c] = __builtin_amdgcn_wmma_f32_16x16x32_f16(
                    false, a[r], false, bf[c], (short)0, acc[r][c], false, false);
    }

    // epilogue: + bias, scatter into q/k/v [B,H,T,D] as f16
    const int row_off = (lane < 16) ? 0 : 8;
#pragma unroll
    for (int r = 0; r < 2; ++r) {
#pragma unroll
        for (int c = 0; c < 2; ++c) {
            const int n_g = n0 + wn + c * 16 + lane16;
            const float bv = bias[n_g];
            const int sec = n_g >> 10;          // 0=q,1=k,2=v
            const int nc  = n_g & 1023;
            const int hh  = nc >> 6;
            const int dd  = nc & 63;
            half_t* dst = (sec == 0) ? qws : (sec == 1) ? kws : vws;
#pragma unroll
            for (int g = 0; g < 8; ++g) {
                const int m_g = m0 + wm + r * 16 + g + row_off;
                const int bidx = m_g >> 11;     // / 2048
                const int t    = m_g & 2047;
                dst[(((size_t)(bidx * H_DIM + hh) * T_DIM + t) * D_DIM) + dd] =
                    (half_t)(acc[r][c][g] + bv);
            }
        }
    }
}

// ---------------------------------------------------------------------------
// Kernel 2: causal flash attention per (b,h). 256 threads = 8 waves,
// 128 q rows per block (16 per wave). KV tiles of 32 keys; online softmax.
// Q and K tiles staged via global_load_async_to_lds_b128 (ASYNCcnt).
// ---------------------------------------------------------------------------
__global__ __launch_bounds__(256) void attn_kernel(
    const half_t* __restrict__ qws, const half_t* __restrict__ kws,
    const half_t* __restrict__ vws, float* __restrict__ out)
{
    __shared__ __attribute__((aligned(64))) half_t Ql[128 * 64];   // [q][d]
    __shared__ __attribute__((aligned(64))) half_t Kl[32 * 64];    // [key][d]
    __shared__ __attribute__((aligned(64))) half_t Vtl[64 * 32];   // [d][key]
    __shared__ __attribute__((aligned(64))) half_t Pl[8 * 16 * 32];// per-wave P

    const int tid  = threadIdx.x;
    const int lane = tid & 31;
    const int wave = tid >> 5;
    const int lane16 = lane & 15;
    const int bh = blockIdx.x;                  // b*16 + h
    const int qb = blockIdx.y * 128;
    const size_t headbase = (size_t)bh * T_DIM * D_DIM;

    // --- stage Q block via async copy to LDS: 4 x b128 per thread ---
#pragma unroll
    for (int it = 0; it < 4; ++it) {
        const int i = (tid + it * 256) * 8;     // half index
        const uint32_t loff = lds_offset(&Ql[i]);
        const half_t* gp = qws + headbase + (size_t)qb * D_DIM + i;
        asm volatile("global_load_async_to_lds_b128 %0, %1, off"
                     :: "v"(loff), "v"(gp) : "memory");
    }
    asm volatile("s_wait_asynccnt 0" ::: "memory");
    __syncthreads();

    const int kbaseA = (lane < 16) ? 0 : 16;
    const int kloB   = (lane < 16) ? 0 : 16;
    const int row_off = (lane < 16) ? 0 : 8;

    // Q fragments (held across KV loop): 16 rows x 64 d = 2 A-fragments
    v16h aq[2];
#pragma unroll
    for (int c = 0; c < 2; ++c) {
        const half_t* ap = &Ql[(wave * 16 + lane16) * 64 + c * 32 + kbaseA];
        v8h lo = *(const v8h*)ap;
        v8h hi = *(const v8h*)(ap + 8);
#pragma unroll
        for (int i = 0; i < 8; ++i) { aq[c][i] = lo[i]; aq[c][i + 8] = hi[i]; }
    }

    v8f o[4] = {};
    float mrow[8], lrow[8];
#pragma unroll
    for (int g = 0; g < 8; ++g) { mrow[g] = -3.0e38f; lrow[g] = 0.0f; }

    const int nkv = (qb >> 5) + 4;              // uniform trip count per block
    for (int j = 0; j < nkv; ++j) {
        const int kvb = j * 32;
        __syncthreads();
        // --- K tile: one async b128 per thread (2048 halves total) ---
        {
            const int i = tid * 8;
            const uint32_t loff = lds_offset(&Kl[i]);
            const half_t* gp = kws + headbase + (size_t)kvb * D_DIM + i;
            asm volatile("global_load_async_to_lds_b128 %0, %1, off"
                         :: "v"(loff), "v"(gp) : "memory");
        }
        // --- V tile: b128 load + transposed b16 stores ---
        {
            const int i = tid * 8;
            const v8h vv = *(const v8h*)(vws + headbase + (size_t)kvb * D_DIM + i);
            const int key = i >> 6, d0 = i & 63;
#pragma unroll
            for (int jj = 0; jj < 8; ++jj)
                Vtl[(d0 + jj) * 32 + key] = vv[jj];
        }
        asm volatile("s_wait_asynccnt 0" ::: "memory");
        __syncthreads();

        // S = Q K^T : 2 key-tiles x 2 d-chunks
        v8f s[2] = {};
#pragma unroll
        for (int tk = 0; tk < 2; ++tk)
#pragma unroll
            for (int c = 0; c < 2; ++c) {
                v16h bk = *(const v16h*)&Kl[(tk * 16 + lane16) * 64 + c * 32 + kloB];
                s[tk] = __builtin_amdgcn_wmma_f32_16x16x32_f16(
                    false, aq[c], false, bk, (short)0, s[tk], false, false);
            }

        // scale, causal mask, online softmax (row reductions within 16-lane group)
        float p0[8], p1[8];
#pragma unroll
        for (int g = 0; g < 8; ++g) {
            const int qg = qb + wave * 16 + g + row_off;
            float v0 = s[0][g] * 0.125f;
            float v1 = s[1][g] * 0.125f;
            if (kvb + lane16 > qg)      v0 = -3.0e38f;
            if (kvb + 16 + lane16 > qg) v1 = -3.0e38f;
            float mx = fmaxf(v0, v1);
#pragma unroll
            for (int d = 1; d < 16; d <<= 1) mx = fmaxf(mx, __shfl_xor(mx, d, 32));
            const float mn = fmaxf(mrow[g], mx);
            const float alpha = __expf(mrow[g] - mn);
            p0[g] = __expf(v0 - mn);
            p1[g] = __expf(v1 - mn);
            float rs = p0[g] + p1[g];
#pragma unroll
            for (int d = 1; d < 16; d <<= 1) rs += __shfl_xor(rs, d, 32);
            lrow[g] = lrow[g] * alpha + rs;
            mrow[g] = mn;
#pragma unroll
            for (int t = 0; t < 4; ++t) o[t][g] *= alpha;
        }

        // convert P (C-layout f32) -> A-layout f16 via per-wave LDS round trip
        half_t* pw = &Pl[wave * 512];
#pragma unroll
        for (int g = 0; g < 8; ++g) {
            const int rr = g + row_off;
            pw[rr * 32 + lane16]      = (half_t)p0[g];
            pw[rr * 32 + 16 + lane16] = (half_t)p1[g];
        }
        __builtin_amdgcn_wave_barrier();
        v16h ap;
        {
            const half_t* q = &pw[lane16 * 32 + kbaseA];
            v8h lo = *(const v8h*)q;
            v8h hi = *(const v8h*)(q + 8);
#pragma unroll
            for (int i = 0; i < 8; ++i) { ap[i] = lo[i]; ap[i + 8] = hi[i]; }
        }
        __builtin_amdgcn_wave_barrier();

        // O += P V : 4 d-tiles
#pragma unroll
        for (int t = 0; t < 4; ++t) {
            v16h bv = *(const v16h*)&Vtl[(t * 16 + lane16) * 32 + kloB];
            o[t] = __builtin_amdgcn_wmma_f32_16x16x32_f16(
                false, ap, false, bv, (short)0, o[t], false, false);
        }
    }

    // epilogue: normalize and write [B,T,C] f32
    const int b_ = bh >> 4, h_ = bh & 15;
#pragma unroll
    for (int t = 0; t < 4; ++t)
#pragma unroll
        for (int g = 0; g < 8; ++g) {
            const int qg = qb + wave * 16 + g + row_off;
            const int dd = t * 16 + lane16;
            out[((size_t)(b_ * T_DIM + qg)) * C_DIM + h_ * D_DIM + dd] =
                o[t][g] / lrow[g];
        }
}

extern "C" void kernel_launch(void* const* d_in, const int* in_sizes, int n_in,
                              void* d_out, int out_size, void* d_ws, size_t ws_size,
                              hipStream_t stream) {
    const float* x    = (const float*)d_in[0];   // [4,2048,1024]
    const float* W    = (const float*)d_in[1];   // [1024,3072]
    const float* bias = (const float*)d_in[2];   // [3072]
    float* out = (float*)d_out;                  // [4,2048,1024]

    const size_t head_elems = (size_t)B_DIM * H_DIM * T_DIM * D_DIM; // 8M
    half_t* qws = (half_t*)d_ws;
    half_t* kws = qws + head_elems;
    half_t* vws = kws + head_elems;              // total 48 MB f16

    dim3 g1(M_DIM / 64, N3C / 128);              // 128 x 24
    qkv_gemm_kernel<<<g1, 256, 0, stream>>>(x, W, bias, qws, kws, vws);

    dim3 g2(B_DIM * H_DIM, T_DIM / 128);         // 64 x 16
    attn_kernel<<<g2, 256, 0, stream>>>(qws, kws, vws, out);
}
